// QNet_25872882991387
// MI455X (gfx1250) — compile-verified, matrix operand order checked by reading, not run on hardware
//
#include <hip/hip_runtime.h>
#include <hip/hip_bf16.h>

// MI455X / gfx1250. wave32. WMMA bf16 16x16x32, f32 accumulate.
//
// Pipeline:
//   0) cvt_weights    : one-shot f32 -> bf16 conversion of W1/W2/Wd into ws
//   1) encode_kernel  : x = relu(relu(obs@W1^T+b1)@W2^T+b2) + comm_t -> ws (bf16)
//                       M=32 per block; obs streamed with non-temporal loads
//   2) lstm_scan      : single persistent workgroup, 8192 sequential steps.
//                       Critical path per step is only: barrier -> ds_load h ->
//                       2 dependent WMMAs per gate -> tanh/elementwise -> store.
//                       The x-part (bias + x@Wih^T) is folded into a pipelined
//                       xacc register tile one step ahead, off the h-dependency
//                       chain. Recurrent weights resident in VGPRs, h double-
//                       buffered in LDS, hidden streamed with coalesced 16B
//                       stores.
//   3) decode_kernel  : q = hidden@Wd^T + bd (f32, non-temporal stores)
//
// Workspace layout (needs ~34.3 MB):
//   [ 0,16MB)     x      bf16 (B*N*H)
//   [16,32MB)     hidden bf16 (B*N*H)
//   [32MB..]      W1/W2/Wd bf16 (2MB + 256KB + 32KB)

typedef __attribute__((ext_vector_type(16))) __bf16 v16bf_t;
typedef __attribute__((ext_vector_type(8)))  float  v8f_t;
typedef __attribute__((ext_vector_type(4)))  float  f4_t;

#define BB   8192
#define NN   16
#define NOBS 512
#define HH   64
#define AA   16
#define H1   128

__device__ __forceinline__ float fast_tanh(float x) {
#if __has_builtin(__builtin_amdgcn_tanhf)
  return __builtin_amdgcn_tanhf(x);
#else
  return tanhf(x);
#endif
}
__device__ __forceinline__ float sigf(float x) {
  return 0.5f * fast_tanh(0.5f * x) + 0.5f;   // sigmoid via hardware tanh
}

// K index of first element of the bf16 pair in VGPR slot v (ISA 7.12.2):
// v<4 -> K=2v, v>=4 -> K=16+2(v-4), plus 8 for the upper lane half.
// Net effect per lane: two contiguous 16B chunks -> compiler emits b128 loads.
__device__ __forceinline__ int kpair_base(int v, int half) {
  return ((v < 4) ? (2 * v) : (8 + 2 * v)) + 8 * half;
}

__device__ __forceinline__ v16bf_t pack16_f32(const float* __restrict__ base,
                                              int rstride, int row, int half, int k0) {
  v16bf_t out;
#pragma unroll
  for (int v = 0; v < 8; ++v) {
    int k = k0 + kpair_base(v, half);
    const float* p = base + (size_t)row * rstride + k;
    out[2 * v]     = (__bf16)p[0];
    out[2 * v + 1] = (__bf16)p[1];
  }
  return out;
}

__device__ __forceinline__ v16bf_t pack16_bf16(const __bf16* __restrict__ base,
                                               int rstride, int row, int half, int k0) {
  v16bf_t out;
#pragma unroll
  for (int v = 0; v < 8; ++v) {
    int k = k0 + kpair_base(v, half);
    const __bf16* p = base + (size_t)row * rstride + k;
    out[2 * v]     = p[0];
    out[2 * v + 1] = p[1];
  }
  return out;
}

__device__ __forceinline__ v8f_t wmma_bf16(v16bf_t a, v16bf_t b, v8f_t c) {
  return __builtin_amdgcn_wmma_f32_16x16x32_bf16(false, a, false, b,
                                                 (short)0, c, false, false);
}

// ---------------------------------------------------------------------------
// Phase 0: one-shot weight conversion f32 -> bf16 into workspace.
// ---------------------------------------------------------------------------
__global__ void __launch_bounds__(256) cvt_weights_kernel(
    const float* __restrict__ W1, const float* __restrict__ W2,
    const float* __restrict__ Wd,
    __bf16* __restrict__ W1b, __bf16* __restrict__ W2b, __bf16* __restrict__ Wdb)
{
  const int stride = gridDim.x * 256;
  const int i0 = blockIdx.x * 256 + threadIdx.x;
  for (int i = i0; i < NN * H1 * NOBS; i += stride) W1b[i] = (__bf16)W1[i];
  for (int i = i0; i < NN * HH * H1;  i += stride) W2b[i] = (__bf16)W2[i];
  for (int i = i0; i < NN * AA * HH;  i += stride) Wdb[i] = (__bf16)Wd[i];
}

// ---------------------------------------------------------------------------
// Phase 1: per-(32-row batch tile, agent) encode MLP. 128 threads = 4 waves.
// grid.x = (B/32) * N = 4096 blocks.
// ---------------------------------------------------------------------------
__global__ void __launch_bounds__(128) encode_kernel(
    const float* __restrict__ obs, const __bf16* __restrict__ W1b,
    const float* __restrict__ b1,  const __bf16* __restrict__ W2b,
    const float* __restrict__ b2,  const float* __restrict__ comm,
    __bf16* __restrict__ xbuf)
{
  __shared__ __align__(16) __bf16 s_obs[32 * NOBS];  // 32 KB
  __shared__ __align__(16) __bf16 s_e1[32 * H1];     // 8 KB

  const int tid  = threadIdx.x;
  const int lane = tid & 31;
  const int wv   = tid >> 5;            // wave 0..3
  const int half = lane >> 4;
  const int j    = lane & 15;

  const int agent = blockIdx.x & 15;
  const int b0    = (blockIdx.x >> 4) * 32;

  // Stage obs tile (32 batch rows x 512) -> LDS bf16. obs is a 256MB
  // read-once stream: non-temporal loads keep it from evicting the L2
  // working set (xbuf/weights) the scan depends on.
  const float* obs_base = obs + ((size_t)b0 * NN + agent) * NOBS;
  for (int i = tid * 4; i < 32 * NOBS; i += 128 * 4) {
    int row = i >> 9, k = i & (NOBS - 1);
    const f4_t v = __builtin_nontemporal_load(
        (const f4_t*)(obs_base + (size_t)row * (NN * NOBS) + k));
    s_obs[row * NOBS + k + 0] = (__bf16)v.x;
    s_obs[row * NOBS + k + 1] = (__bf16)v.y;
    s_obs[row * NOBS + k + 2] = (__bf16)v.z;
    s_obs[row * NOBS + k + 3] = (__bf16)v.w;
  }
  __syncthreads();

  // GEMM1: e1 = relu(obs @ W1[n]^T + b1[n]).
  // Wave wv owns cols [32wv, 32wv+32) x 2 M-tiles; B operands shared over M.
  const __bf16* W1n = W1b + (size_t)agent * H1 * NOBS;
  v8f_t acc1[2][2];   // [mtile][coltile]
#pragma unroll
  for (int t = 0; t < 2; ++t) {
    const float bv = b1[agent * H1 + 32 * wv + 16 * t + j];
#pragma unroll
    for (int r = 0; r < 8; ++r) { acc1[0][t][r] = bv; acc1[1][t][r] = bv; }
  }
  for (int ks = 0; ks < NOBS / 32; ++ks) {
    v16bf_t a0  = pack16_bf16(s_obs, NOBS, j, half, 32 * ks);
    v16bf_t a1  = pack16_bf16(s_obs + 16 * NOBS, NOBS, j, half, 32 * ks);
    v16bf_t bb0 = pack16_bf16(W1n, NOBS, 32 * wv + j, half, 32 * ks);
    v16bf_t bb1 = pack16_bf16(W1n, NOBS, 32 * wv + 16 + j, half, 32 * ks);
    acc1[0][0] = wmma_bf16(a0, bb0, acc1[0][0]);
    acc1[0][1] = wmma_bf16(a0, bb1, acc1[0][1]);
    acc1[1][0] = wmma_bf16(a1, bb0, acc1[1][0]);
    acc1[1][1] = wmma_bf16(a1, bb1, acc1[1][1]);
  }
#pragma unroll
  for (int mt = 0; mt < 2; ++mt)
#pragma unroll
    for (int t = 0; t < 2; ++t)
#pragma unroll
      for (int r = 0; r < 8; ++r) {
        float x = acc1[mt][t][r] > 0.f ? acc1[mt][t][r] : 0.f;
        int m = mt * 16 + r + 8 * half;
        s_e1[m * H1 + 32 * wv + 16 * t + j] = (__bf16)x;
      }
  __syncthreads();

  // GEMM2: x = relu(e1 @ W2[n]^T + b2[n]) + comm_t[n].
  // Wave wv owns cols [16wv,16wv+16) x 2 M-tiles.
  {
    const int c0 = 16 * wv;
    const __bf16* W2n = W2b + (size_t)agent * HH * H1;
    v8f_t acc2[2];
    const float bv = b2[agent * HH + c0 + j];
#pragma unroll
    for (int r = 0; r < 8; ++r) { acc2[0][r] = bv; acc2[1][r] = bv; }
#pragma unroll
    for (int ks = 0; ks < H1 / 32; ++ks) {
      v16bf_t bb = pack16_bf16(W2n, H1, c0 + j, half, 32 * ks);
      v16bf_t a0 = pack16_bf16(s_e1, H1, j, half, 32 * ks);
      v16bf_t a1 = pack16_bf16(s_e1 + 16 * H1, H1, j, half, 32 * ks);
      acc2[0] = wmma_bf16(a0, bb, acc2[0]);
      acc2[1] = wmma_bf16(a1, bb, acc2[1]);
    }
    const float cv = comm[(c0 + j) * NN + agent];   // comm[0,0,h,n] -> comm_t[n,h]
#pragma unroll
    for (int mt = 0; mt < 2; ++mt)
#pragma unroll
      for (int r = 0; r < 8; ++r) {
        float x = (acc2[mt][r] > 0.f ? acc2[mt][r] : 0.f) + cv;
        int m = mt * 16 + r + 8 * half;
        xbuf[((size_t)(b0 + m) * NN + agent) * HH + c0 + j] = (__bf16)x;
      }
  }
}

// ---------------------------------------------------------------------------
// Phase 2: sequential LSTM scan, single persistent workgroup of 4 waves.
// Wave wv owns h-column slice [16wv,16wv+16) for all four gates (i,f,g,o);
// cell state, gate accumulators, and the pipelined x-part never leave VGPRs.
// ---------------------------------------------------------------------------
__global__ void __launch_bounds__(128) lstm_scan_kernel(
    const __bf16* __restrict__ xbuf,
    const float* __restrict__ Wih, const float* __restrict__ Whh,
    const float* __restrict__ bih, const float* __restrict__ bhh,
    const float* __restrict__ h0,  const float* __restrict__ c0s,
    __bf16* __restrict__ hidden)
{
  __shared__ __align__(16) __bf16 s_h[2][16 * HH];   // double-buffered h, 2x2KB

  const int tid  = threadIdx.x;
  const int lane = tid & 31;
  const int wv   = tid >> 5;
  const int half = lane >> 4;
  const int j    = lane & 15;

  // Recurrent weights resident in VGPRs: 4 gate tiles x 2 K-chunks, x/h parts.
  v16bf_t Bx[4][2], Bh[4][2];
  v8f_t   bias_acc[4];        // C-operand seed: D = A*B + bias
#pragma unroll
  for (int g = 0; g < 4; ++g) {
    const int col = 64 * g + 16 * wv + j;        // torch gate order i,f,g,o
    const float bv = bih[col] + bhh[col];
#pragma unroll
    for (int r = 0; r < 8; ++r) bias_acc[g][r] = bv;
#pragma unroll
    for (int kc = 0; kc < 2; ++kc) {
      Bx[g][kc] = pack16_f32(Wih, HH, col, half, 32 * kc);
      Bh[g][kc] = pack16_f32(Whh, HH, col, half, 32 * kc);
    }
  }

  // Initial cell state: lstm_s is (H,N); c tile element (m, 16wv+j).
  v8f_t c;
#pragma unroll
  for (int r = 0; r < 8; ++r) {
    int m = r + 8 * half;
    c[r] = c0s[(16 * wv + j) * NN + m];
  }
  // Initial h -> LDS buffer 1 (step 0 reads prev = 1).
  for (int i = tid; i < 16 * HH; i += 128) {
    int n = i >> 6, k = i & (HH - 1);
    s_h[1][i] = (__bf16)h0[k * NN + n];
  }
  __syncthreads();

  // Pipelined x-part: xacc[g] = bias + x_b @ Wih^T, computed one step ahead
  // so it is OFF the h-dependency chain.
  v8f_t xacc[4];
  {
    v16bf_t ax0 = pack16_bf16(xbuf, HH, j, half, 0);
    v16bf_t ax1 = pack16_bf16(xbuf, HH, j, half, 32);
#pragma unroll
    for (int g = 0; g < 4; ++g) {
      v8f_t a = wmma_bf16(ax0, Bx[g][0], bias_acc[g]);
      xacc[g] = wmma_bf16(ax1, Bx[g][1], a);
    }
  }

  for (int b = 0; b < BB; ++b) {
    const int cur = b & 1, prev = cur ^ 1;

    // Critical path: h(b-1) from LDS -> only 2 dependent WMMAs per gate.
    v16bf_t ah0 = pack16_bf16(s_h[prev], HH, j, half, 0);
    v16bf_t ah1 = pack16_bf16(s_h[prev], HH, j, half, 32);

    // Off-path: stream h(b-1) to global, coalesced 16B per thread.
    if (b > 0) {
      ((uint4*)(hidden + (size_t)(b - 1) * (NN * HH)))[tid] =
          ((const uint4*)s_h[prev])[tid];
    }

    v8f_t acc[4];
#pragma unroll
    for (int g = 0; g < 4; ++g) {
      v8f_t a = wmma_bf16(ah0, Bh[g][0], xacc[g]);
      acc[g]  = wmma_bf16(ah1, Bh[g][1], a);
    }

#pragma unroll
    for (int r = 0; r < 8; ++r) {
      float iv = sigf(acc[0][r]);
      float fv = sigf(acc[1][r]);
      float gv = fast_tanh(acc[2][r]);
      float ov = sigf(acc[3][r]);
      float cn = fv * c[r] + iv * gv;
      c[r] = cn;
      float hv = ov * fast_tanh(cn);
      int m = r + 8 * half;
      s_h[cur][m * HH + 16 * wv + j] = (__bf16)hv;
    }

    // Off-path: fold x(b+1) into xacc for the next step (wrap; branch-free).
    {
      const __bf16* xn = xbuf + (size_t)((b + 1) & (BB - 1)) * (NN * HH);
      v16bf_t nx0 = pack16_bf16(xn, HH, j, half, 0);
      v16bf_t nx1 = pack16_bf16(xn, HH, j, half, 32);
#pragma unroll
      for (int g = 0; g < 4; ++g) {
        v8f_t a = wmma_bf16(nx0, Bx[g][0], bias_acc[g]);
        xacc[g] = wmma_bf16(nx1, Bx[g][1], a);
      }
    }

    __syncthreads();   // writes of buf `cur` visible before step b+1 reads it
  }
  // Final step's h.
  ((uint4*)(hidden + (size_t)(BB - 1) * (NN * HH)))[tid] =
      ((const uint4*)s_h[(BB - 1) & 1])[tid];
}

// ---------------------------------------------------------------------------
// Phase 3: q = hidden @ Wd[n]^T + bd[n]. 8 waves/block, 1 tile/wave, 2 WMMAs.
// grid = B/16*N/8 = 1024 blocks.
// ---------------------------------------------------------------------------
__global__ void __launch_bounds__(256) decode_kernel(
    const __bf16* __restrict__ hidden, const __bf16* __restrict__ Wdb,
    const float* __restrict__ bd, float* __restrict__ q)
{
  const int tid  = threadIdx.x;
  const int lane = tid & 31;
  const int wv   = tid >> 5;
  const int half = lane >> 4;
  const int j    = lane & 15;

  const int gid   = blockIdx.x * 8 + wv;
  const int agent = gid & 15;
  const int b0    = (gid >> 4) * 16;

  const __bf16* hbase = hidden + ((size_t)b0 * NN + agent) * HH;  // row stride N*H
  const __bf16* Wdn   = Wdb + (size_t)agent * AA * HH;

  v8f_t acc;
  const float bv = bd[agent * AA + j];
#pragma unroll
  for (int r = 0; r < 8; ++r) acc[r] = bv;

  v16bf_t a0 = pack16_bf16(hbase, NN * HH, j, half, 0);
  v16bf_t a1 = pack16_bf16(hbase, NN * HH, j, half, 32);
  v16bf_t w0 = pack16_bf16(Wdn, HH, j, half, 0);
  v16bf_t w1 = pack16_bf16(Wdn, HH, j, half, 32);
  acc = wmma_bf16(a0, w0, acc);
  acc = wmma_bf16(a1, w1, acc);

  // q is final output, never re-read on device: non-temporal stores.
#pragma unroll
  for (int r = 0; r < 8; ++r) {
    int m = r + 8 * half;
    __builtin_nontemporal_store(acc[r],
        q + ((size_t)(b0 + m) * NN + agent) * AA + j);
  }
}

extern "C" void kernel_launch(void* const* d_in, const int* in_sizes, int n_in,
                              void* d_out, int out_size, void* d_ws, size_t ws_size,
                              hipStream_t stream) {
  const float* obs  = (const float*)d_in[0];
  const float* lh   = (const float*)d_in[1];   // lstm_h (1,1,H,N)
  const float* ls   = (const float*)d_in[2];   // lstm_s (1,1,H,N)
  const float* comm = (const float*)d_in[3];   // (1,1,H,N)
  const float* W1   = (const float*)d_in[4];
  const float* b1   = (const float*)d_in[5];
  const float* W2   = (const float*)d_in[6];
  const float* b2   = (const float*)d_in[7];
  const float* Wih  = (const float*)d_in[8];
  const float* Whh  = (const float*)d_in[9];
  const float* bih  = (const float*)d_in[10];
  const float* bhh  = (const float*)d_in[11];
  const float* Wd   = (const float*)d_in[12];
  const float* bd   = (const float*)d_in[13];

  __bf16* xbuf   = (__bf16*)d_ws;                          // 16 MB
  __bf16* hidden = xbuf + (size_t)BB * NN * HH;            // +16 MB
  __bf16* W1b    = hidden + (size_t)BB * NN * HH;          // +2 MB
  __bf16* W2b    = W1b + (size_t)NN * H1 * NOBS;           // +256 KB
  __bf16* Wdb    = W2b + (size_t)NN * HH * H1;             // +32 KB
  float*  q      = (float*)d_out;

  cvt_weights_kernel<<<512, 256, 0, stream>>>(W1, W2, Wd, W1b, W2b, Wdb);
  encode_kernel<<<(BB / 32) * NN, 128, 0, stream>>>(obs, W1b, b1, W2b, b2, comm, xbuf);
  lstm_scan_kernel<<<1, 128, 0, stream>>>(xbuf, Wih, Whh, bih, bhh, lh, ls, hidden);
  decode_kernel<<<(BB / 16) * NN / 8, 256, 0, stream>>>(hidden, Wdb, bd, q);
}
